// GIN_34316788695392
// MI455X (gfx1250) — matrix operand order repaired and checked by reading, not compiled, not run on hardware
//
#include <hip/hip_runtime.h>

#define NN 50000
#define DD 128
#define NE 1600000

typedef __attribute__((ext_vector_type(16))) __bf16 v16bf;
typedef __attribute__((ext_vector_type(8)))  float  v8f;

__device__ __forceinline__ __bf16 f2bf(float f) {
  union { float f; unsigned u; } v; v.f = f;
  unsigned r = v.u + 0x7FFFu + ((v.u >> 16) & 1u);   // round-to-nearest-even
  unsigned short h = (unsigned short)(r >> 16);
  union { unsigned short s; __bf16 b; } o; o.s = h;
  return o.b;
}

// ---------------- Phase 1: scatter-add aggregation --------------------------
// One wave (32 lanes) per edge. Lane l handles channels [4l, 4l+4):
// coalesced float4 gather of the 512B source row + 4 hardware f32 atomics.
// x and agg are 25.6 MB each -> resident in the 192 MB L2; this phase is
// L2 atomic-throughput bound, with only ~90 MB of true HBM traffic.
__global__ __launch_bounds__(256) void gin_scatter(
    const float* __restrict__ x,
    const int*   __restrict__ src,
    const int*   __restrict__ dst,
    float*       __restrict__ agg)
{
  long long gid = (long long)blockIdx.x * blockDim.x + threadIdx.x;
  int edge = (int)(gid >> 5);
  if (edge >= NE) return;
  int l = (int)(gid & 31);
  int s = src[edge];
  int d = dst[edge];
  const float4 v = *reinterpret_cast<const float4*>(x + (long long)s * DD + l * 4);
  float* p = agg + (long long)d * DD + l * 4;
  unsafeAtomicAdd(p + 0, v.x);
  unsafeAtomicAdd(p + 1, v.y);
  unsafeAtomicAdd(p + 2, v.z);
  unsafeAtomicAdd(p + 3, v.w);
}

// ---------------- Phase 2: fused (x+agg) -> Linear -> ReLU -> Linear --------
// 256 threads = 8 waves; block covers 128 node rows (16 rows/wave).
// Dynamic LDS layout (97 KB):
//   [0,32768)       : Wt (128x128 bf16, TRANSPOSED: Wt[n*128+k]), W1 then W2
//   [32768,33280)   : bias (128 f32)
//   [33280,66048)   : hA: 8 x (16x128 bf16) per-wave input tiles
//   [66048,98816)   : hB: 8 x (16x128 bf16) per-wave ReLU tiles
extern __shared__ char gin_smem[];

// A tile (16x32 bf16), row-major source with stride DD.
// Lane<16:  row=lane,    e0..15 -> K=k0+0..15   (2x b128)
// Lane>=16: row=lane-16, e0..15 -> K=k0+16..31  (2x b128)
__device__ __forceinline__ v16bf load_a_tile(const __bf16* tile, int lane, int k0) {
  v16bf a;
  int row  = lane & 15;
  int aoff = (lane >> 4) << 4;         // 0 or 16
  const __bf16* p = tile + row * DD + k0 + aoff;
#pragma unroll
  for (int e = 0; e < 8; ++e)  a[e] = p[e];
#pragma unroll
  for (int e = 8; e < 16; ++e) a[e] = p[e];
  return a;
}

// B tile (32x16 bf16) from TRANSPOSED weights Wt[n*128+k]:
// lane holds column n = n0+(lane&15); K = k0 + (lane>=16?16:0) + e,
// i.e. 16 contiguous bf16 per lane -> two ds_load_b128.
__device__ __forceinline__ v16bf load_b_tile(const __bf16* Wt, int lane, int k0, int n0) {
  v16bf b;
  int col = n0 + (lane & 15);
  int kb  = k0 + ((lane >> 4) << 4);   // 0 or 16
  const __bf16* p = Wt + col * DD + kb;
#pragma unroll
  for (int e = 0; e < 16; ++e) b[e] = p[e];
  return b;
}

// One 16x128 row-strip GEMM: out16x128 = A(16x128) * W(128x128), K pipelined.
// All 8 tile-loads of an nt-iteration are issued before the 4 WMMAs so the
// scheduler can overlap tail loads with early WMMAs (partial dscnt waits).
template <bool RELU_TO_LDS>
__device__ __forceinline__ void strip_gemm(
    const __bf16* __restrict__ aTile,   // 16x128 bf16 (LDS, wave-private)
    const __bf16* __restrict__ Wt,      // 128x128 bf16 transposed (LDS)
    const float*  __restrict__ bias,    // 128 f32 (LDS)
    int lane,
    __bf16* __restrict__ reluOut,       // used when RELU_TO_LDS
    float*  __restrict__ gOut,          // used when !RELU_TO_LDS
    int rowBase)
{
  // Hoist the 8 bias scalars this lane will need (one per nt).
  float bv[8];
#pragma unroll
  for (int nt = 0; nt < 8; ++nt) bv[nt] = bias[nt * 16 + (lane & 15)];

#pragma unroll 1
  for (int nt = 0; nt < 8; ++nt) {
    v16bf a0 = load_a_tile(aTile, lane, 0);
    v16bf b0 = load_b_tile(Wt, lane, 0, nt * 16);
    v16bf a1 = load_a_tile(aTile, lane, 32);
    v16bf b1 = load_b_tile(Wt, lane, 32, nt * 16);
    v16bf a2 = load_a_tile(aTile, lane, 64);
    v16bf b2 = load_b_tile(Wt, lane, 64, nt * 16);
    v16bf a3 = load_a_tile(aTile, lane, 96);
    v16bf b3 = load_b_tile(Wt, lane, 96, nt * 16);
    v8f acc = {};
    acc = __builtin_amdgcn_wmma_f32_16x16x32_bf16(false, a0, false, b0, (short)0, acc, false, false);
    acc = __builtin_amdgcn_wmma_f32_16x16x32_bf16(false, a1, false, b1, (short)0, acc, false, false);
    acc = __builtin_amdgcn_wmma_f32_16x16x32_bf16(false, a2, false, b2, (short)0, acc, false, false);
    acc = __builtin_amdgcn_wmma_f32_16x16x32_bf16(false, a3, false, b3, (short)0, acc, false, false);

    int ncol = nt * 16 + (lane & 15);
    if (RELU_TO_LDS) {
#pragma unroll
      for (int r2 = 0; r2 < 8; ++r2) {
        int m = r2 + ((lane >> 4) << 3);
        float v = acc[r2] + bv[nt];
        reluOut[m * DD + ncol] = f2bf(v > 0.f ? v : 0.f);
      }
    } else {
#pragma unroll
      for (int r2 = 0; r2 < 8; ++r2) {
        int m = r2 + ((lane >> 4) << 3);
        int g = rowBase + m;
        if (g < NN) gOut[(long long)g * DD + ncol] = acc[r2] + bv[nt];
      }
    }
  }
}

__global__ __launch_bounds__(256) void gin_mlp(
    const float* __restrict__ x,  const float* __restrict__ agg,
    const float* __restrict__ W1, const float* __restrict__ b1,
    const float* __restrict__ W2, const float* __restrict__ b2,
    float* __restrict__ out)
{
  __bf16* Wt   = (__bf16*)gin_smem;
  float*  bias = (float*)(gin_smem + 32768);
  __bf16* hA   = (__bf16*)(gin_smem + 33280);
  __bf16* hB   = (__bf16*)(gin_smem + 66048);

  const int tid  = threadIdx.x;
  const int wave = tid >> 5;
  const int lane = tid & 31;
  const int rowBase = blockIdx.x * 128 + wave * 16;
  __bf16* myA = hA + wave * (16 * DD);
  __bf16* myB = hB + wave * (16 * DD);

  // Stage h = (1+eps)*x + agg  (eps == 0) as bf16, wave-private tile.
#pragma unroll 4
  for (int i = 0; i < 64; ++i) {
    int idx = lane + i * 32;
    int r = idx >> 7, c = idx & 127;
    int g = rowBase + r;
    float v = 0.f;
    if (g < NN) v = x[(long long)g * DD + c] + agg[(long long)g * DD + c];
    myA[idx] = f2bf(v);
  }
  // Stage W1 transposed (bf16) and b1. Global read coalesced; LDS store
  // scattered (once per block, amortized over 8 tile-loads per wave).
#pragma unroll 4
  for (int i = 0; i < 64; ++i) {
    int idx = tid + i * 256;            // idx = k*128 + n in W1
    int k = idx >> 7, n = idx & 127;
    Wt[n * DD + k] = f2bf(W1[idx]);
  }
  if (tid < DD) bias[tid] = b1[tid];
  __syncthreads();

  // GEMM1 + bias + ReLU -> hB
  strip_gemm<true>(myA, Wt, bias, lane, myB, nullptr, rowBase);
  __syncthreads();

  // Swap in W2 / b2 (reuse LDS weight buffer).
#pragma unroll 4
  for (int i = 0; i < 64; ++i) {
    int idx = tid + i * 256;
    int k = idx >> 7, n = idx & 127;
    Wt[n * DD + k] = f2bf(W2[idx]);
  }
  if (tid < DD) bias[tid] = b2[tid];
  __syncthreads();

  // GEMM2 + bias -> out (f32)
  strip_gemm<false>(myB, Wt, bias, lane, nullptr, out, rowBase);
}

// ---------------------------------------------------------------------------
extern "C" void kernel_launch(void* const* d_in, const int* in_sizes, int n_in,
                              void* d_out, int out_size, void* d_ws, size_t ws_size,
                              hipStream_t stream) {
  const float* x  = (const float*)d_in[0];
  const int*   ei = (const int*)d_in[1];      // [2, NE]: row0 = src, row1 = dst
  const float* W1 = (const float*)d_in[2];
  const float* b1 = (const float*)d_in[3];
  const float* W2 = (const float*)d_in[4];
  const float* b2 = (const float*)d_in[5];
  float* out = (float*)d_out;
  float* agg = (float*)d_ws;                  // NN*DD floats = 25.6 MB scratch

  hipMemsetAsync(agg, 0, (size_t)NN * DD * sizeof(float), stream);

  long long total = (long long)NE * 32;       // one wave per edge
  int sblocks = (int)((total + 255) / 256);
  gin_scatter<<<sblocks, 256, 0, stream>>>(x, ei, ei + NE, agg);

  int mblocks = (NN + 127) / 128;
  size_t shmem = 33280 + 2 * 32768;           // 98816 B dynamic LDS
  gin_mlp<<<mblocks, 256, shmem, stream>>>(x, agg, W1, b1, W2, b2, out);
}
